// multi_head_self_attention_87273735454825
// MI455X (gfx1250) — matrix operand order repaired and checked by reading, not compiled, not run on hardware
//
#include <hip/hip_runtime.h>
#include <hip/hip_bf16.h>

#define D_MODEL 1024
#define N_HEAD  16
#define D_K     64
#define SEQ_LEN 2048
#define BATCH   2

typedef __attribute__((ext_vector_type(16))) _Float16 v16h;
typedef __attribute__((ext_vector_type(8)))  _Float16 v8h;
typedef __attribute__((ext_vector_type(4)))  _Float16 v4h;
typedef __attribute__((ext_vector_type(8)))  float    v8f;

static __device__ __forceinline__ v8h load8(const _Float16* p) {
    return *(const v8h*)p;
}
static __device__ __forceinline__ v16h mk16(v8h lo, v8h hi) {
    v16h r;
#pragma unroll
    for (int i = 0; i < 8; ++i) { r[i] = lo[i]; r[i + 8] = hi[i]; }
    return r;
}
static __device__ __forceinline__ v8f wmma_f16(v16h a, v16h b, v8f c) {
    return __builtin_amdgcn_wmma_f32_16x16x32_f16(false, a, false, b,
                                                  (short)0, c, false, false);
}

// ---------------------------------------------------------------------------
// Stage 0: fp32 -> f16 conversion (grid-stride, 4 elements/thread/iter)
// ---------------------------------------------------------------------------
__global__ void cvt_f32_to_f16_kernel(const float4* __restrict__ src,
                                      v4h* __restrict__ dst, int n4) {
    int i = blockIdx.x * blockDim.x + threadIdx.x;
    int stride = gridDim.x * blockDim.x;
    for (; i < n4; i += stride) {
        float4 v = src[i];
        v4h h;
        h[0] = (_Float16)v.x; h[1] = (_Float16)v.y;
        h[2] = (_Float16)v.z; h[3] = (_Float16)v.w;
        dst[i] = h;
    }
}

// ---------------------------------------------------------------------------
// Stage 1: projection  Y = X @ W^T + b, scattered to per-head layout.
// One wave per 64x64 output tile: 4 row-subtiles reuse each W fragment
// (16 WMMAs per k-step against 8 fragment loads -> 4x the L2 intensity).
// transpose_out == 0: out[((b*H+h)*S + s)*64 + d]   (Q, K)
// transpose_out == 1: out[((b*H+h)*64 + d)*S + s]   (V^T)
// ---------------------------------------------------------------------------
__global__ __launch_bounds__(32)
void qkv_proj_kernel(const _Float16* __restrict__ X,
                     const _Float16* __restrict__ W,
                     const float* __restrict__ bias,
                     _Float16* __restrict__ out, int transpose_out) {
    const int lane  = threadIdx.x;
    const int ln    = lane & 15;
    const int abase = (lane < 16) ? 0 : 8;    // A-fragment k base
    const int bbase = (lane < 16) ? 0 : 8;    // B-fragment k base
    const int mrow  = (lane < 16) ? 0 : 8;    // C/D row offset per VGPR
    const int mt = blockIdx.x;                // 64-row tile of M = B*S
    const int nt = blockIdx.y;                // 64-col tile of N = D

    v8f acc[4][4] = {};                       // [m-subtile][n-subtile]
    for (int kc = 0; kc < D_MODEL; kc += 32) {
        v16h a[4];
#pragma unroll
        for (int mi = 0; mi < 4; ++mi) {
            const _Float16* xp =
                X + (size_t)(mt * 64 + mi * 16 + ln) * D_MODEL + kc;
            a[mi] = mk16(load8(xp + abase), load8(xp + abase + 16));
        }
#pragma unroll
        for (int nd = 0; nd < 4; ++nd) {
            const int n = nt * 64 + nd * 16 + ln;
            const _Float16* wp = W + (size_t)n * D_MODEL + kc + bbase;
            v16h bfrag = mk16(load8(wp), load8(wp + 16));
#pragma unroll
            for (int mi = 0; mi < 4; ++mi)
                acc[mi][nd] = wmma_f16(a[mi], bfrag, acc[mi][nd]);
        }
    }

#pragma unroll
    for (int nd = 0; nd < 4; ++nd) {
        const int f  = nt * 64 + nd * 16 + ln;   // feature index in [0,1024)
        const float bv = bias[f];
        const int h  = f >> 6;
        const int dd = f & 63;
#pragma unroll
        for (int mi = 0; mi < 4; ++mi) {
#pragma unroll
            for (int v = 0; v < 8; ++v) {
                const int t = mt * 64 + mi * 16 + v + mrow;  // token in [0,B*S)
                const int b = t >> 11;                        // / SEQ_LEN
                const int s = t & (SEQ_LEN - 1);
                const float val = acc[mi][nd][v] + bv;
                size_t off;
                if (transpose_out)
                    off = ((size_t)((b * N_HEAD + h) * D_K + dd)) * SEQ_LEN + s;
                else
                    off = ((size_t)((b * N_HEAD + h) * SEQ_LEN + s)) * D_K + dd;
                out[off] = (_Float16)val;
            }
        }
    }
}

// ---------------------------------------------------------------------------
// Stage 2: flash attention. One wave per (b, h, 32-query tile): two 16-row
// query subtiles share every K/V fragment (2x L2 intensity).
// Q,K: [B,H,S,64] f16.  Vt: [B,H,64,S] f16.  mask: [B,S] bytes (1 = masked).
// Online softmax over 32-key tiles; P transposed C-layout -> A-layout via LDS.
// ---------------------------------------------------------------------------
__global__ __launch_bounds__(32)
void flash_attn_kernel(const _Float16* __restrict__ Q,
                       const _Float16* __restrict__ K,
                       const _Float16* __restrict__ Vt,
                       const unsigned char* __restrict__ maskb,
                       float* __restrict__ out) {
    __shared__ __align__(16) _Float16 sP[16 * 40];  // 16x32 tile, stride 40

    const int lane  = threadIdx.x;
    const int ln    = lane & 15;
    const int abase = (lane < 16) ? 0 : 8;
    const int bbase = (lane < 16) ? 0 : 8;
    const int mrow  = (lane < 16) ? 0 : 8;

    const int qt0 = blockIdx.x * 2;   // first 16-query subtile index
    const int h   = blockIdx.y;
    const int b   = blockIdx.z;
    const size_t bh = (size_t)(b * N_HEAD + h);

    const _Float16* Qb = Q  + bh * SEQ_LEN * D_K;
    const _Float16* Kb = K  + bh * SEQ_LEN * D_K;
    const _Float16* Vb = Vt + bh * D_K * SEQ_LEN;
    const unsigned char* mb = maskb + (size_t)b * SEQ_LEN;

    // Q fragments (A layout), two query subtiles x two k=32 chunks
    v16h qa[2][2];
#pragma unroll
    for (int qt = 0; qt < 2; ++qt)
#pragma unroll
        for (int c = 0; c < 2; ++c) {
            const _Float16* p =
                Qb + (size_t)((qt0 + qt) * 16 + ln) * D_K + c * 32;
            qa[qt][c] = mk16(load8(p + abase), load8(p + abase + 16));
        }

    float mi[2][8], li[2][8];
#pragma unroll
    for (int qt = 0; qt < 2; ++qt)
#pragma unroll
        for (int v = 0; v < 8; ++v) { mi[qt][v] = -1.0e30f; li[qt][v] = 0.0f; }
    v8f o[2][4] = {};

    const float SC = 0.18033688f;  // (1/sqrt(64)) * log2(e)

    for (int kv0 = 0; kv0 < SEQ_LEN; kv0 += 32) {
        // ---- scores: S = Q @ K^T; K fragments reused by both query subtiles
        v8f sf[2][2];
#pragma unroll
        for (int nsub = 0; nsub < 2; ++nsub) {
            const _Float16* kp = Kb + (size_t)(kv0 + nsub * 16 + ln) * D_K;
            v16h k0 = mk16(load8(kp + bbase),      load8(kp + bbase + 16));
            v16h k1 = mk16(load8(kp + 32 + bbase), load8(kp + 32 + bbase + 16));
#pragma unroll
            for (int qt = 0; qt < 2; ++qt) {
                v8f z = {};
                z = wmma_f16(qa[qt][0], k0, z);
                sf[qt][nsub] = wmma_f16(qa[qt][1], k1, z);
            }
        }

        // ---- V fragments (B layout over kv), shared by both query subtiles
        v16h vfrag[4];
#pragma unroll
        for (int nd = 0; nd < 4; ++nd) {
            const _Float16* vp =
                Vb + (size_t)(nd * 16 + ln) * SEQ_LEN + kv0 + bbase;
            vfrag[nd] = mk16(load8(vp), load8(vp + 16));
        }

        if (kv0 + 32 < SEQ_LEN)
            __builtin_prefetch(Kb + (size_t)(kv0 + 32 + ln) * D_K, 0, 0);

        // ---- key mask (lane = column = key index)
        const bool mk0 = mb[kv0 + ln] != 0;
        const bool mk1 = mb[kv0 + 16 + ln] != 0;

#pragma unroll
        for (int qt = 0; qt < 2; ++qt) {
            // ---- online softmax per C/D-layout row
            float alpha[8];
#pragma unroll
            for (int v = 0; v < 8; ++v) {
                float z0 = mk0 ? -1.0e30f : sf[qt][0][v] * SC;
                float z1 = mk1 ? -1.0e30f : sf[qt][1][v] * SC;
                float rm = fmaxf(z0, z1);
                rm = fmaxf(rm, __shfl_xor(rm, 1, 16));
                rm = fmaxf(rm, __shfl_xor(rm, 2, 16));
                rm = fmaxf(rm, __shfl_xor(rm, 4, 16));
                rm = fmaxf(rm, __shfl_xor(rm, 8, 16));
                float mn = fmaxf(mi[qt][v], rm);
                float a  = exp2f(mi[qt][v] - mn);
                float p0 = exp2f(z0 - mn);
                float p1 = exp2f(z1 - mn);
                float rs = p0 + p1;
                rs += __shfl_xor(rs, 1, 16);
                rs += __shfl_xor(rs, 2, 16);
                rs += __shfl_xor(rs, 4, 16);
                rs += __shfl_xor(rs, 8, 16);
                li[qt][v] = li[qt][v] * a + rs;
                mi[qt][v] = mn;
                alpha[v] = a;
                // stage P (C layout -> LDS, row-major 16x32, stride 40)
                const int r = v + mrow;
                sP[r * 40 + ln]      = (_Float16)p0;
                sP[r * 40 + 16 + ln] = (_Float16)p1;
            }

            // rescale accumulator
#pragma unroll
            for (int nd = 0; nd < 4; ++nd)
#pragma unroll
                for (int v = 0; v < 8; ++v) o[qt][nd][v] *= alpha[v];

            // ---- reload P in A layout from LDS (same-wave LDS is in order)
            v16h pa = mk16(*(const v8h*)&sP[ln * 40 + abase],
                           *(const v8h*)&sP[ln * 40 + abase + 16]);

            // ---- O += P @ V
#pragma unroll
            for (int nd = 0; nd < 4; ++nd)
                o[qt][nd] = wmma_f16(pa, vfrag[nd], o[qt][nd]);
        }
    }

    // ---- epilogue: out[b, s, h*64+d] = O / l
#pragma unroll
    for (int qt = 0; qt < 2; ++qt)
#pragma unroll
        for (int nd = 0; nd < 4; ++nd) {
            const int f = h * 64 + nd * 16 + ln;
#pragma unroll
            for (int v = 0; v < 8; ++v) {
                const int s = (qt0 + qt) * 16 + v + mrow;
                out[((size_t)(b * SEQ_LEN + s)) * D_MODEL + f] =
                    o[qt][nd][v] / li[qt][v];
            }
        }
}

// ---------------------------------------------------------------------------
// Host launcher
// ---------------------------------------------------------------------------
extern "C" void kernel_launch(void* const* d_in, const int* in_sizes, int n_in,
                              void* d_out, int out_size, void* d_ws, size_t ws_size,
                              hipStream_t stream) {
    (void)in_sizes; (void)n_in; (void)out_size; (void)ws_size;

    const float*         x    = (const float*)d_in[0];
    const unsigned char* mask = (const unsigned char*)d_in[1];  // jax bool = 1 byte
    const float* Wq = (const float*)d_in[2];
    const float* bq = (const float*)d_in[3];
    const float* Wk = (const float*)d_in[4];
    const float* bk = (const float*)d_in[5];
    const float* Wv = (const float*)d_in[6];
    const float* bv = (const float*)d_in[7];
    float* out = (float*)d_out;

    // workspace layout (f16 buffers)
    const size_t nX = (size_t)BATCH * SEQ_LEN * D_MODEL;        // 4,194,304
    const size_t nW = (size_t)D_MODEL * D_MODEL;                // 1,048,576
    const size_t nQ = (size_t)BATCH * N_HEAD * SEQ_LEN * D_K;   // 4,194,304
    char* w = (char*)d_ws;
    _Float16* Xh  = (_Float16*)(w);
    _Float16* Wqh = (_Float16*)(w + 2 * nX);
    _Float16* Wkh = Wqh + nW;
    _Float16* Wvh = Wkh + nW;
    _Float16* Qh  = Wvh + nW;
    _Float16* Kh  = Qh + nQ;
    _Float16* Vth = Kh + nQ;

    // Stage 0: convert activations + weights to f16
    cvt_f32_to_f16_kernel<<<dim3(1024), dim3(256), 0, stream>>>(
        (const float4*)x, (v4h*)Xh, (int)(nX / 4));
    cvt_f32_to_f16_kernel<<<dim3(512), dim3(256), 0, stream>>>(
        (const float4*)Wq, (v4h*)Wqh, (int)(nW / 4));
    cvt_f32_to_f16_kernel<<<dim3(512), dim3(256), 0, stream>>>(
        (const float4*)Wk, (v4h*)Wkh, (int)(nW / 4));
    cvt_f32_to_f16_kernel<<<dim3(512), dim3(256), 0, stream>>>(
        (const float4*)Wv, (v4h*)Wvh, (int)(nW / 4));

    // Stage 1: Q, K ([B,H,S,64]) and V^T ([B,H,64,S]) projections
    dim3 pgrid((BATCH * SEQ_LEN) / 64, D_MODEL / 64);
    qkv_proj_kernel<<<pgrid, dim3(32), 0, stream>>>(Xh, Wqh, bq, Qh, 0);
    qkv_proj_kernel<<<pgrid, dim3(32), 0, stream>>>(Xh, Wkh, bk, Kh, 0);
    qkv_proj_kernel<<<pgrid, dim3(32), 0, stream>>>(Xh, Wvh, bv, Vth, 1);

    // Stage 2: flash attention (32 queries per wave)
    dim3 agrid(SEQ_LEN / 32, N_HEAD, BATCH);
    flash_attn_kernel<<<agrid, dim3(32), 0, stream>>>(Qh, Kh, Vth, mask, out);
}
